// DCN_21749714387188
// MI455X (gfx1250) — matrix operand order repaired and checked by reading, not compiled, not run on hardware
//
#include <hip/hip_runtime.h>
#include <hip/hip_bf16.h>
#include <stdint.h>

typedef _Float16 v16h __attribute__((ext_vector_type(16)));
typedef _Float16 v8h  __attribute__((ext_vector_type(8)));
typedef float    v8f  __attribute__((ext_vector_type(8)));

static constexpr int BB = 16384;   // batch
static constexpr int FF = 26;      // fields
static constexpr int EE = 32;      // embedding dim
static constexpr int DD = FF * EE; // 832
static constexpr int N0 = 1024, N1 = 512, N2 = 256;

// ---------------------------------------------------------------------------
// Embedding gather: x0[b, f*32+e] = table[idx[b,f]*32+e] * val[b,f]
// writes both fp32 (for cross branch) and f16 (for WMMA deep branch)
// ---------------------------------------------------------------------------
__global__ __launch_bounds__(256)
void gather_embed(const int* __restrict__ fidx, const float* __restrict__ fval,
                  const float* __restrict__ table,
                  float* __restrict__ x0, _Float16* __restrict__ x0h) {
  size_t idx = (size_t)blockIdx.x * blockDim.x + threadIdx.x;  // exact grid: B*D
  int b = (int)(idx / DD);
  int j = (int)(idx % DD);
  int f = j >> 5, e = j & 31;
  int fi = fidx[b * FF + f];
  float v = table[(size_t)fi * EE + e] * fval[b * FF + f];
  x0[idx]  = v;
  x0h[idx] = (_Float16)v;
}

// ---------------------------------------------------------------------------
// Weight repack: W[K][N] fp32 -> Wt[N][K] f16 (row-major over K, contiguous)
// ---------------------------------------------------------------------------
__global__ __launch_bounds__(256)
void transpose_to_f16(const float* __restrict__ W, _Float16* __restrict__ Wt,
                      int K, int N) {
  size_t idx = (size_t)blockIdx.x * blockDim.x + threadIdx.x;
  if (idx >= (size_t)K * N) return;
  int n = (int)(idx / K);
  int k = (int)(idx % K);
  Wt[idx] = (_Float16)W[(size_t)k * N + n];
}

// ---------------------------------------------------------------------------
// WMMA fragment loaders (wave32, 16x16x32 f16)
// A 16x32 layout (ISA 7.12.2): lanes 0-15 row M=lane hold K {0-7,16-23};
//                              lanes 16-31 same rows hold K {8-15,24-31}.
// B 32x16 layout (matches A): lanes 0-15 col N=lane hold K {0-7,16-23};
//                             lanes 16-31 hold K {8-15,24-31}.
// ---------------------------------------------------------------------------
__device__ __forceinline__ v16h load_a_frag(const _Float16* __restrict__ A,
                                            int ldk, int m0, int k0, int lane) {
  int r  = lane & 15;
  int kb = (lane & 16) ? 8 : 0;
  const _Float16* p = A + (size_t)(m0 + r) * ldk + (k0 + kb);
  v8h lo = *(const v8h*)p;          // K = kb .. kb+7
  v8h hi = *(const v8h*)(p + 16);   // K = kb+16 .. kb+23
  return __builtin_shufflevector(lo, hi, 0,1,2,3,4,5,6,7,8,9,10,11,12,13,14,15);
}

__device__ __forceinline__ v16h load_b_frag(const _Float16* __restrict__ Wt,
                                            int ldk, int n0, int k0, int lane) {
  int c  = lane & 15;
  int kb = (lane & 16) ? 8 : 0;
  const _Float16* p = Wt + (size_t)(n0 + c) * ldk + (k0 + kb);
  v8h lo = *(const v8h*)p;          // K = kb .. kb+7
  v8h hi = *(const v8h*)(p + 16);   // K = kb+16 .. kb+23
  return __builtin_shufflevector(lo, hi, 0,1,2,3,4,5,6,7,8,9,10,11,12,13,14,15);
}

// ---------------------------------------------------------------------------
// C = relu(A[M,K] @ Wt[N,K]^T + bias)   A,Wt,C: f16, acc f32 via WMMA
// block = 256 threads = 8 waves arranged 2(M) x 4(N)
// wave tile 64(M) x 64(N) -> 16 accumulators, 16 wmma per 16 b128 loads
// block tile 128(M) x 256(N); K multiple of 32
// bias is folded into the initial C accumulator (D = A.B + C chain).
// ---------------------------------------------------------------------------
__global__ __launch_bounds__(256)
void gemm_relu_wmma(const _Float16* __restrict__ A, const _Float16* __restrict__ Wt,
                    const float* __restrict__ bias, _Float16* __restrict__ C,
                    int M, int N, int K) {
  const int lane = threadIdx.x & 31;
  const int wave = threadIdx.x >> 5;
  const int m0 = blockIdx.y * 128 + (wave >> 2) * 64;
  const int n0 = blockIdx.x * 256 + (wave & 3) * 64;

  const int nl = lane & 15;           // N within a 16-wide tile (C/D layout)
  const int mo = (lane & 16) ? 8 : 0; // M offset for upper lane half

  v8f acc[4][4];
#pragma unroll
  for (int j = 0; j < 4; j++) {
    float bv = bias[n0 + j * 16 + nl];
    v8f binit = {bv, bv, bv, bv, bv, bv, bv, bv};
#pragma unroll
    for (int i = 0; i < 4; i++) acc[i][j] = binit;
  }

  for (int k0 = 0; k0 < K; k0 += 32) {
    v16h a[4], b[4];
#pragma unroll
    for (int i = 0; i < 4; i++) a[i] = load_a_frag(A, K, m0 + i * 16, k0, lane);
#pragma unroll
    for (int j = 0; j < 4; j++) b[j] = load_b_frag(Wt, K, n0 + j * 16, k0, lane);
#pragma unroll
    for (int i = 0; i < 4; i++)
#pragma unroll
      for (int j = 0; j < 4; j++)
        acc[i][j] = __builtin_amdgcn_wmma_f32_16x16x32_f16(
            false, a[i], false, b[j], (short)0, acc[i][j], false, false);
  }

  // C/D layout: vgpr r -> M = r (lanes 0-15) or r+8 (lanes 16-31); N = lane&15
#pragma unroll
  for (int i = 0; i < 4; i++) {
#pragma unroll
    for (int j = 0; j < 4; j++) {
      int n = n0 + j * 16 + nl;
#pragma unroll
      for (int r = 0; r < 8; r++) {
        float v = acc[i][j][r];
        v = v > 0.0f ? v : 0.0f;          // relu
        int m = m0 + i * 16 + mo + r;
        C[(size_t)m * N + n] = (_Float16)v;
      }
    }
  }
}

// ---------------------------------------------------------------------------
// Cross branch (3 layers, all in registers, one wave per row) fused with the
// cross half of the output projection:  out[b] = y_cross . out_w[:832] + out_b
// ---------------------------------------------------------------------------
__global__ __launch_bounds__(256)
void cross_out(const float* __restrict__ x0, const float* __restrict__ cw,
               const float* __restrict__ cb, const float* __restrict__ ow,
               const float* __restrict__ ob, float* __restrict__ out) {
  const int lane = threadIdx.x & 31;
  const int row  = blockIdx.x * 8 + (threadIdx.x >> 5);
  const float* xr = x0 + (size_t)row * DD;
  float x[FF], y[FF];
#pragma unroll
  for (int t = 0; t < FF; t++) { x[t] = xr[lane + 32 * t]; y[t] = x[t]; }
#pragma unroll
  for (int i = 0; i < 3; i++) {
    float p = 0.0f;
#pragma unroll
    for (int t = 0; t < FF; t++) p += y[t] * cw[i * DD + lane + 32 * t];
#pragma unroll
    for (int off = 16; off > 0; off >>= 1) p += __shfl_xor(p, off, 32);
    float bb = cb[i];
#pragma unroll
    for (int t = 0; t < FF; t++) y[t] = x[t] * p + bb + y[t];
  }
  float p = 0.0f;
#pragma unroll
  for (int t = 0; t < FF; t++) p += y[t] * ow[lane + 32 * t];
#pragma unroll
  for (int off = 16; off > 0; off >>= 1) p += __shfl_xor(p, off, 32);
  if (lane == 0) out[row] = p + ob[0];
}

// out[b] += y_deep . out_w[832:1088]
__global__ __launch_bounds__(256)
void deep_out(const _Float16* __restrict__ h2, const float* __restrict__ ow,
              float* __restrict__ out) {
  const int lane = threadIdx.x & 31;
  const int row  = blockIdx.x * 8 + (threadIdx.x >> 5);
  const _Float16* hr = h2 + (size_t)row * N2;
  float p = 0.0f;
#pragma unroll
  for (int t = 0; t < 8; t++)
    p += (float)hr[lane + 32 * t] * ow[DD + lane + 32 * t];
#pragma unroll
  for (int off = 16; off > 0; off >>= 1) p += __shfl_xor(p, off, 32);
  if (lane == 0) out[row] += p;
}

// ---------------------------------------------------------------------------
extern "C" void kernel_launch(void* const* d_in, const int* in_sizes, int n_in,
                              void* d_out, int out_size, void* d_ws, size_t ws_size,
                              hipStream_t stream) {
  (void)in_sizes; (void)n_in; (void)out_size; (void)ws_size;
  const int*   fidx  = (const int*)  d_in[0];
  const float* fval  = (const float*)d_in[1];
  // d_in[2] numeric_value (width 0), d_in[3] label: unused
  const float* table = (const float*)d_in[4];
  const float* cw    = (const float*)d_in[5];
  const float* cb    = (const float*)d_in[6];
  const float* w0    = (const float*)d_in[7];
  const float* b0    = (const float*)d_in[8];
  const float* w1    = (const float*)d_in[9];
  const float* b1    = (const float*)d_in[10];
  const float* w2    = (const float*)d_in[11];
  const float* b2    = (const float*)d_in[12];
  const float* ow    = (const float*)d_in[13];
  const float* ob    = (const float*)d_in[14];
  float* out = (float*)d_out;

  uint8_t* base = (uint8_t*)d_ws;
  size_t off = 0;
  auto carve = [&](size_t bytes) -> uint8_t* {
    uint8_t* p = base + off;
    off = (off + bytes + 255) & ~(size_t)255;
    return p;
  };
  float*    x0  = (float*)   carve((size_t)BB * DD * 4);
  _Float16* x0h = (_Float16*)carve((size_t)BB * DD * 2);
  _Float16* h0  = (_Float16*)carve((size_t)BB * N0 * 2);
  _Float16* h1  = (_Float16*)carve((size_t)BB * N1 * 2);
  _Float16* h2  = (_Float16*)carve((size_t)BB * N2 * 2);
  _Float16* wt0 = (_Float16*)carve((size_t)N0 * DD * 2);
  _Float16* wt1 = (_Float16*)carve((size_t)N1 * N0 * 2);
  _Float16* wt2 = (_Float16*)carve((size_t)N2 * N1 * 2);

  gather_embed<<<(BB * (size_t)DD) / 256, 256, 0, stream>>>(fidx, fval, table, x0, x0h);

  transpose_to_f16<<<((size_t)N0 * DD + 255) / 256, 256, 0, stream>>>(w0, wt0, DD, N0);
  transpose_to_f16<<<((size_t)N1 * N0 + 255) / 256, 256, 0, stream>>>(w1, wt1, N0, N1);
  transpose_to_f16<<<((size_t)N2 * N1 + 255) / 256, 256, 0, stream>>>(w2, wt2, N1, N2);

  gemm_relu_wmma<<<dim3(N0 / 256, BB / 128), 256, 0, stream>>>(x0h, wt0, b0, h0, BB, N0, DD);
  gemm_relu_wmma<<<dim3(N1 / 256, BB / 128), 256, 0, stream>>>(h0,  wt1, b1, h1, BB, N1, N0);
  gemm_relu_wmma<<<dim3(N2 / 256, BB / 128), 256, 0, stream>>>(h1,  wt2, b2, h2, BB, N2, N1);

  cross_out<<<BB / 8, 256, 0, stream>>>(x0, cw, cb, ow, ob, out);
  deep_out <<<BB / 8, 256, 0, stream>>>(h2, ow, out);
}
